// GateRetention_69861938037189
// MI455X (gfx1250) — matrix operand order repaired and checked by reading, not compiled, not run on hardware
//
#include <hip/hip_runtime.h>
#include <hip/hip_bf16.h>

typedef __attribute__((ext_vector_type(16))) __bf16 v16bf;
typedef __attribute__((ext_vector_type(8)))  __bf16 v8bf;
typedef __attribute__((ext_vector_type(8)))  float  v8f;

constexpr int   Hh    = 16;
constexpr int   HD    = 128;
constexpr int   CHUNK = 256;
constexpr int   NCH   = 16;
constexpr int   Tt    = 4096;
constexpr int   Bb    = 2;
constexpr int   Dd    = 2048;
constexpr int   BT    = Bb * Tt;          // 8192
constexpr float GLNf  = 16.0f;
constexpr float EPSf  = 1e-5f;
constexpr float SCALE = 0.0883883476483184405f;   // 128^-0.5
constexpr float INV_SCALE = 11.3137084989847604f; // 128^0.5

__device__ inline float ldf(const float* p)  { return *p; }
__device__ inline float ldf(const __bf16* p) { return (float)*p; }
__device__ inline void  stc(float* p, float v)  { *p = v; }
__device__ inline void  stc(__bf16* p, float v) { *p = (__bf16)v; }

__device__ inline v16bf join16(v8bf lo, v8bf hi) {
    return __builtin_shufflevector(lo, hi, 0, 1, 2, 3, 4, 5, 6, 7,
                                            8, 9, 10, 11, 12, 13, 14, 15);
}

// Async byte-copy global -> LDS (16B), tracked by ASYNCcnt.
__device__ inline void async_b128(unsigned ldsOff, const void* gaddr) {
    asm volatile("global_load_async_to_lds_b128 %0, %1, off"
                 :: "v"(ldsOff), "v"((unsigned long long)(size_t)gaddr)
                 : "memory");
}
__device__ inline void wait_async0() {
    asm volatile("s_wait_asynccnt 0" ::: "memory");
}

// ---------------------------------------------------------------------------
// Tiled GEMM: C[M,N] = A[M,K] (f32 or bf16) @ W[K,N] (f32), via bf16 WMMA.
// Block tile 128x128, 8 waves, wave tile 64x32 (4x2 of 16x16), K-step 32.
// ---------------------------------------------------------------------------
template <typename TIN, typename TOUT>
__global__ __launch_bounds__(256)
void gemm_wmma(const TIN* __restrict__ A, const float* __restrict__ W,
               TOUT* __restrict__ C, int M, int N, int K)
{
    __shared__ __align__(16) __bf16 As[128][40];   // 128x32 used (+pad, 80B rows)
    __shared__ __align__(16) __bf16 Bs[32][136];   // 32x128 used (+pad)

    const int tid  = threadIdx.x;
    const int lane = tid & 31;
    const int wid  = tid >> 5;
    const int r    = lane & 15;
    const int hi   = lane >> 4;
    const int wM   = wid & 1;        // 2 waves along M (64 rows each)
    const int wN   = wid >> 1;       // 4 waves along N (32 cols each)
    const long mBase = (long)blockIdx.y * 128;
    const int  nBase = blockIdx.x * 128;

    v8f acc[4][2] = {};

    const int arow = tid >> 1, ac0 = (tid & 1) * 16;   // A-tile loader coords
    const int brow = tid >> 3, bc0 = (tid & 7) * 16;   // B-tile loader coords

    #pragma unroll 2
    for (int kt = 0; kt < K; kt += 32) {
        {
            const TIN* ap = A + (mBase + arow) * (long)K + kt + ac0;
            #pragma unroll
            for (int j = 0; j < 16; ++j) As[arow][ac0 + j] = (__bf16)ldf(&ap[j]);
            const float* wp = W + (long)(kt + brow) * N + nBase + bc0;
            #pragma unroll
            for (int j = 0; j < 16; ++j) Bs[brow][bc0 + j] = (__bf16)wp[j];
        }
        __syncthreads();

        if (kt + 32 < K) {   // speculative prefetch of next tiles (global_prefetch_b8)
            __builtin_prefetch(A + (mBase + arow) * (long)K + kt + 32 + ac0, 0, 1);
            __builtin_prefetch(W + (long)(kt + 32 + brow) * N + nBase + bc0, 0, 1);
        }

        v16bf af[4], bfrag[2];
        #pragma unroll
        for (int mt = 0; mt < 4; ++mt) {
            const int mr = wM * 64 + mt * 16 + r;
            v8bf lo = *(const v8bf*)&As[mr][hi * 8];
            v8bf h8 = *(const v8bf*)&As[mr][16 + hi * 8];
            af[mt] = join16(lo, h8);
        }
        #pragma unroll
        for (int nt = 0; nt < 2; ++nt) {
            const int nc = wN * 32 + nt * 16 + r;
            #pragma unroll
            for (int j = 0; j < 16; ++j) bfrag[nt][j] = Bs[(j < 8 ? hi * 8 + j : 8 + hi * 8 + j)][nc];
        }
        #pragma unroll
        for (int mt = 0; mt < 4; ++mt)
            #pragma unroll
            for (int nt = 0; nt < 2; ++nt)
                acc[mt][nt] = __builtin_amdgcn_wmma_f32_16x16x32_bf16(
                    false, af[mt], false, bfrag[nt], (short)0, acc[mt][nt],
                    false, false);
        __syncthreads();
    }

    #pragma unroll
    for (int mt = 0; mt < 4; ++mt)
        #pragma unroll
        for (int nt = 0; nt < 2; ++nt)
            #pragma unroll
            for (int rr = 0; rr < 8; ++rr) {
                const long row = mBase + wM * 64 + mt * 16 + rr + hi * 8;
                const int  col = nBase + wN * 32 + nt * 16 + r;
                stc(C + row * (long)N + col, acc[mt][nt][rr]);
            }
}

// ---------------------------------------------------------------------------
// gt = log_sigmoid(x @ Wgt) / 16.  N = H = 16 -> one WMMA N-tile per wave.
// ---------------------------------------------------------------------------
__global__ __launch_bounds__(32)
void gate_proj_kernel(const float* __restrict__ x, const float* __restrict__ Wgt,
                      float* __restrict__ gt)
{
    const int lane = threadIdx.x & 31;
    const int r = lane & 15, hi = lane >> 4;
    const long m0 = (long)blockIdx.x * 16;

    v8f acc = {};
    #pragma unroll 4
    for (int kk = 0; kk < Dd / 32; ++kk) {
        v16bf a, b;
        const float* xp = x + (m0 + r) * (long)Dd + kk * 32;
        #pragma unroll
        for (int j = 0; j < 8; ++j) {
            a[j]     = (__bf16)xp[hi * 8 + j];
            a[j + 8] = (__bf16)xp[16 + hi * 8 + j];
        }
        #pragma unroll
        for (int j = 0; j < 16; ++j)
            b[j] = (__bf16)Wgt[(long)(kk * 32 + (j < 8 ? hi * 8 + j : 8 + hi * 8 + j)) * Hh + r];
        acc = __builtin_amdgcn_wmma_f32_16x16x32_bf16(false, a, false, b,
                                                      (short)0, acc, false, false);
    }
    #pragma unroll
    for (int rr = 0; rr < 8; ++rr) {
        const long row = m0 + rr + hi * 8;
        const float v  = acc[rr];
        const float ls = fminf(v, 0.0f) - __logf(1.0f + __expf(-fabsf(v)));
        gt[row * Hh + r] = ls * (1.0f / GLNf);
    }
}

// ---------------------------------------------------------------------------
// Interleaved RoPE on q and k (bf16 in-place, f32 math). The hd^-0.5 scale is
// folded into q here (compensated by 1/scale in q_dec and scale in k_dec).
// ---------------------------------------------------------------------------
__global__ void rope_kernel(__bf16* __restrict__ q, __bf16* __restrict__ k,
                            const float* __restrict__ cosb,
                            const float* __restrict__ sinb)
{
    const long idx = (long)blockIdx.x * blockDim.x + threadIdx.x; // BT*H*64
    if (idx >= (long)BT * Hh * (HD / 2)) return;
    const int  i  = (int)(idx & 63);
    const int  h  = (int)((idx >> 6) & 15);
    const long bt = idx >> 10;
    const int  t  = (int)(bt % Tt);
    const float c = cosb[t * 64 + i], s = sinb[t * 64 + i];
    const long base = bt * Dd + h * HD + 2 * i;
    float x1 = (float)q[base], x2 = (float)q[base + 1];
    q[base]     = (__bf16)((x1 * c - x2 * s) * SCALE);
    q[base + 1] = (__bf16)((x1 * s + x2 * c) * SCALE);
    x1 = (float)k[base]; x2 = (float)k[base + 1];
    k[base]     = (__bf16)(x1 * c - x2 * s);
    k[base + 1] = (__bf16)(x1 * s + x2 * c);
}

// ---------------------------------------------------------------------------
// Fused chunked gated retention. One block per (b,h); sequential over chunks.
// k tile and q strips staged via global_load_async_to_lds_b128 (ASYNCcnt),
// overlapped with the gate cumsum scan. LDS ~150KB.
// ---------------------------------------------------------------------------
__global__ __launch_bounds__(256)
void retention_kernel(const __bf16* __restrict__ q, const __bf16* __restrict__ k,
                      const __bf16* __restrict__ v, const __bf16* __restrict__ g,
                      const float* __restrict__ gt, __bf16* __restrict__ o)
{
    __shared__ __align__(16) __bf16 ks[CHUNK][HD];            // raw k (async)
    __shared__ __align__(16) __bf16 qs[64][HD];               // scaled q strip (async)
    __shared__ __align__(16) __bf16 Sm[HD][HD];               // running state
    __shared__ __align__(16) unsigned char scratch[64 * 256 * 2]; // scores / o-buf
    __shared__ float gc[CHUNK];
    __shared__ float kd[CHUNK];
    __shared__ float rs[64][4];

    __bf16 (*sc)[256] = reinterpret_cast<__bf16(*)[256]>(scratch);
    float  (*ob)[HD]  = reinterpret_cast<float(*)[HD]>(scratch);

    const int tid  = threadIdx.x;
    const int lane = tid & 31;
    const int wid  = tid >> 5;
    const int r    = lane & 15;
    const int hi   = lane >> 4;
    const int b    = blockIdx.x / Hh;
    const int h    = blockIdx.x % Hh;

    const unsigned ksBase = (unsigned)(size_t)&ks[0][0];
    const unsigned qsBase = (unsigned)(size_t)&qs[0][0];

    for (int i = tid; i < HD * HD; i += 256) Sm[i >> 7][i & 127] = (__bf16)0.0f;
    __syncthreads();

    for (int n = 0; n < NCH; ++n) {
        const long t0 = (long)b * Tt + n * CHUNK;    // base row in BT space

        // ---- issue async staging of k tile (raw bytes), overlap with scan ----
        #pragma unroll
        for (int it = 0; it < 16; ++it) {
            const int i  = it * 256 + tid;           // 16B chunk id (4096 total)
            const int rr = i >> 4, dd = (i & 15) * 8;
            async_b128(ksBase + (unsigned)(rr * HD + dd) * 2,
                       k + ((t0 + rr) * Hh + h) * HD + dd);
        }

        // ---- gate cumsum (Hillis-Steele) ----
        gc[tid] = gt[(t0 + tid) * Hh + h];
        __syncthreads();
        for (int off = 1; off < CHUNK; off <<= 1) {
            const float a  = gc[tid];
            const float ad = (tid >= off) ? gc[tid - off] : 0.0f;
            __syncthreads();
            gc[tid] = a + ad;
            __syncthreads();
        }
        const float gtot = gc[CHUNK - 1];

        wait_async0();
        __syncthreads();

        // ---- process chunk in 64-row quarters ----
        for (int qr = 0; qr < 4; ++qr) {
            const int r0 = qr * 64;

            // stage the 64-row scaled-q strip asynchronously
            #pragma unroll
            for (int it = 0; it < 4; ++it) {
                const int i  = it * 256 + tid;       // 1024 chunks of 16B
                const int rr = i >> 4, dd = (i & 15) * 8;
                async_b128(qsBase + (unsigned)(rr * HD + dd) * 2,
                           q + ((t0 + r0 + rr) * Hh + h) * HD + dd);
            }
            wait_async0();
            __syncthreads();

            // phase A: scores = (q*scale) @ k^T  (64x256, K=128), decay mask
            {
                const int wM = wid & 1;    // 32 rows each
                const int wN = wid >> 1;   // 64 cols each
                v8f accA[2][4] = {};
                #pragma unroll
                for (int kk = 0; kk < 4; ++kk) {
                    v16bf am[2], bn[4];
                    #pragma unroll
                    for (int mt = 0; mt < 2; ++mt) {
                        const int ml = wM * 32 + mt * 16 + r;
                        v8bf lo = *(const v8bf*)&qs[ml][kk * 32 + hi * 8];
                        v8bf h8 = *(const v8bf*)&qs[ml][kk * 32 + 16 + hi * 8];
                        am[mt] = join16(lo, h8);
                    }
                    #pragma unroll
                    for (int nt = 0; nt < 4; ++nt) {
                        const int s = wN * 64 + nt * 16 + r;
                        #pragma unroll
                        for (int j = 0; j < 16; ++j)
                            bn[nt][j] = ks[s][kk * 32 + (j < 8 ? hi * 8 + j : 8 + hi * 8 + j)];
                    }
                    #pragma unroll
                    for (int mt = 0; mt < 2; ++mt)
                        #pragma unroll
                        for (int nt = 0; nt < 4; ++nt)
                            accA[mt][nt] = __builtin_amdgcn_wmma_f32_16x16x32_bf16(
                                false, am[mt], false, bn[nt], (short)0,
                                accA[mt][nt], false, false);
                }
                #pragma unroll
                for (int mt = 0; mt < 2; ++mt)
                    #pragma unroll
                    for (int nt = 0; nt < 4; ++nt)
                        #pragma unroll
                        for (int rr = 0; rr < 8; ++rr) {
                            const int c = r0 + wM * 32 + mt * 16 + rr + hi * 8;
                            const int s = wN * 64 + nt * 16 + r;
                            float val = accA[mt][nt][rr];
                            val = (c >= s) ? val * __expf(gc[c] - gc[s]) : 0.0f;
                            sc[c - r0][s] = (__bf16)val;
                        }
            }
            __syncthreads();

            // phase B: o = scores @ v  +  (q * q_dec) @ S   (64x128)
            const int wM2 = wid & 1;    // 32 rows
            const int wN2 = wid >> 1;   // 32 cols
            v8f accO[2][2] = {};
            #pragma unroll
            for (int kk = 0; kk < 8; ++kk) {          // K = 256 over s
                v16bf am[2], bn[2];
                #pragma unroll
                for (int mt = 0; mt < 2; ++mt) {
                    const int ml = wM2 * 32 + mt * 16 + r;
                    v8bf lo = *(const v8bf*)&sc[ml][kk * 32 + hi * 8];
                    v8bf h8 = *(const v8bf*)&sc[ml][kk * 32 + 16 + hi * 8];
                    am[mt] = join16(lo, h8);
                }
                #pragma unroll
                for (int nt = 0; nt < 2; ++nt) {
                    const int e = wN2 * 32 + nt * 16 + r;
                    #pragma unroll
                    for (int j = 0; j < 16; ++j) {
                        const int s = kk * 32 + (j < 8 ? hi * 8 + j : 8 + hi * 8 + j);
                        bn[nt][j] = v[((t0 + s) * Hh + h) * HD + e];
                    }
                }
                #pragma unroll
                for (int mt = 0; mt < 2; ++mt)
                    #pragma unroll
                    for (int nt = 0; nt < 2; ++nt)
                        accO[mt][nt] = __builtin_amdgcn_wmma_f32_16x16x32_bf16(
                            false, am[mt], false, bn[nt], (short)0,
                            accO[mt][nt], false, false);
            }
            #pragma unroll
            for (int kk = 0; kk < 4; ++kk) {          // K = 128 over d (cross)
                v16bf am[2], bn[2];
                #pragma unroll
                for (int mt = 0; mt < 2; ++mt) {
                    const int ml = wM2 * 32 + mt * 16 + r;
                    const float qd = __expf(gc[r0 + ml]) * INV_SCALE;
                    #pragma unroll
                    for (int j = 0; j < 8; ++j) {
                        am[mt][j]     = (__bf16)((float)qs[ml][kk * 32 + hi * 8 + j] * qd);
                        am[mt][j + 8] = (__bf16)((float)qs[ml][kk * 32 + 16 + hi * 8 + j] * qd);
                    }
                }
                #pragma unroll
                for (int nt = 0; nt < 2; ++nt) {
                    const int e = wN2 * 32 + nt * 16 + r;
                    #pragma unroll
                    for (int j = 0; j < 16; ++j)
                        bn[nt][j] = Sm[kk * 32 + (j < 8 ? hi * 8 + j : 8 + hi * 8 + j)][e];
                }
                #pragma unroll
                for (int mt = 0; mt < 2; ++mt)
                    #pragma unroll
                    for (int nt = 0; nt < 2; ++nt)
                        accO[mt][nt] = __builtin_amdgcn_wmma_f32_16x16x32_bf16(
                            false, am[mt], false, bn[nt], (short)0,
                            accO[mt][nt], false, false);
            }
            __syncthreads();   // done reading sc -> reuse as ob

            #pragma unroll
            for (int mt = 0; mt < 2; ++mt)
                #pragma unroll
                for (int nt = 0; nt < 2; ++nt)
                    #pragma unroll
                    for (int rr = 0; rr < 8; ++rr) {
                        const int row = wM2 * 32 + mt * 16 + rr + hi * 8;
                        const int col = wN2 * 32 + nt * 16 + r;
                        ob[row][col] = accO[mt][nt][rr];
                    }
            __syncthreads();

            // RMSNorm over hd + silu(g) gate + store
            {
                const int row = tid >> 2, seg = tid & 3;
                float ssum = 0.0f;
                #pragma unroll
                for (int d0 = seg * 32; d0 < seg * 32 + 32; ++d0) {
                    const float t = ob[row][d0];
                    ssum += t * t;
                }
                rs[row][seg] = ssum;
            }
            __syncthreads();
            {
                const int row = tid >> 2, seg = tid & 3;
                const float tot = rs[row][0] + rs[row][1] + rs[row][2] + rs[row][3];
                const float rn  = rsqrtf(tot * (1.0f / HD) + EPSf);
                const long  bt  = t0 + r0 + row;
                #pragma unroll
                for (int d0 = seg * 32; d0 < seg * 32 + 32; ++d0) {
                    const float gv  = (float)g[bt * Dd + h * HD + d0];
                    const float sil = gv / (1.0f + __expf(-gv));
                    o[bt * Dd + h * HD + d0] = (__bf16)(ob[row][d0] * rn * sil);
                }
            }
            __syncthreads();
        }

        // ---- state update: S = S*exp(gtot) + (k*scale*k_dec)^T @ v ----
        kd[tid] = __expf(gtot - gc[tid]) * SCALE;
        __syncthreads();
        {
            const float cdec = __expf(gtot);
            const int wM3 = wid & 1;    // 64 d-rows each
            const int wN3 = wid >> 1;   // 32 e-cols each
            v8f accS[4][2] = {};
            #pragma unroll
            for (int kk = 0; kk < 8; ++kk) {          // K = 256 over s
                v16bf am[4], bn[2];
                #pragma unroll
                for (int mt = 0; mt < 4; ++mt) {
                    const int d = wM3 * 64 + mt * 16 + r;
                    #pragma unroll
                    for (int j = 0; j < 8; ++j) {
                        const int s0 = kk * 32 + hi * 8 + j;
                        const int s1 = kk * 32 + 16 + hi * 8 + j;
                        am[mt][j]     = (__bf16)((float)ks[s0][d] * kd[s0]);
                        am[mt][j + 8] = (__bf16)((float)ks[s1][d] * kd[s1]);
                    }
                }
                #pragma unroll
                for (int nt = 0; nt < 2; ++nt) {
                    const int e = wN3 * 32 + nt * 16 + r;
                    #pragma unroll
                    for (int j = 0; j < 16; ++j) {
                        const int s = kk * 32 + (j < 8 ? hi * 8 + j : 8 + hi * 8 + j);
                        bn[nt][j] = v[((t0 + s) * Hh + h) * HD + e];
                    }
                }
                #pragma unroll
                for (int mt = 0; mt < 4; ++mt)
                    #pragma unroll
                    for (int nt = 0; nt < 2; ++nt)
                        accS[mt][nt] = __builtin_amdgcn_wmma_f32_16x16x32_bf16(
                            false, am[mt], false, bn[nt], (short)0,
                            accS[mt][nt], false, false);
            }
            #pragma unroll
            for (int mt = 0; mt < 4; ++mt)
                #pragma unroll
                for (int nt = 0; nt < 2; ++nt)
                    #pragma unroll
                    for (int rr = 0; rr < 8; ++rr) {
                        const int d = wM3 * 64 + mt * 16 + rr + hi * 8;
                        const int e = wN3 * 32 + nt * 16 + r;
                        Sm[d][e] = (__bf16)((float)Sm[d][e] * cdec + accS[mt][nt][rr]);
                    }
        }
        __syncthreads();
    }
}

// ---------------------------------------------------------------------------
extern "C" void kernel_launch(void* const* d_in, const int* in_sizes, int n_in,
                              void* d_out, int out_size, void* d_ws, size_t ws_size,
                              hipStream_t stream)
{
    const float* x    = (const float*)d_in[0];
    const float* cosb = (const float*)d_in[1];
    const float* sinb = (const float*)d_in[2];
    const float* Wq   = (const float*)d_in[3];
    const float* Wk   = (const float*)d_in[4];
    const float* Wv   = (const float*)d_in[5];
    const float* Wg   = (const float*)d_in[6];
    const float* Wgt  = (const float*)d_in[7];
    const float* Wout = (const float*)d_in[8];
    float* out = (float*)d_out;

    char* ws = (char*)d_ws;
    const size_t sz = (size_t)BT * Dd * sizeof(__bf16);   // 32 MiB each
    __bf16* qw  = (__bf16*)(ws);
    __bf16* kw  = (__bf16*)(ws + sz);
    __bf16* vw  = (__bf16*)(ws + 2 * sz);
    __bf16* gw  = (__bf16*)(ws + 3 * sz);
    __bf16* owb = (__bf16*)(ws + 4 * sz);
    float*  gtw = (float*)(ws + 5 * sz);

    dim3 gg(Dd / 128, BT / 128);
    gemm_wmma<float, __bf16><<<gg, 256, 0, stream>>>(x, Wq, qw, BT, Dd, Dd);
    gemm_wmma<float, __bf16><<<gg, 256, 0, stream>>>(x, Wk, kw, BT, Dd, Dd);
    gemm_wmma<float, __bf16><<<gg, 256, 0, stream>>>(x, Wv, vw, BT, Dd, Dd);
    gemm_wmma<float, __bf16><<<gg, 256, 0, stream>>>(x, Wg, gw, BT, Dd, Dd);
    gate_proj_kernel<<<BT / 16, 32, 0, stream>>>(x, Wgt, gtw);

    const long nrope = (long)BT * Hh * (HD / 2);
    rope_kernel<<<(unsigned)((nrope + 255) / 256), 256, 0, stream>>>(qw, kw, cosb, sinb);

    retention_kernel<<<Bb * Hh, 256, 0, stream>>>(qw, kw, vw, gw, gtw, owb);

    gemm_wmma<__bf16, float><<<gg, 256, 0, stream>>>(owb, Wout, out, BT, Dd, Dd);
}
